// QuantumState_9637906612482
// MI455X (gfx1250) — matrix operand order repaired and checked by reading, not compiled, not run on hardware
//
#include <hip/hip_runtime.h>

// ---------------- problem constants ----------------
#define B_   32
#define L_   4096
#define D_   512
#define NS_  64           // n_states
#define NC_  128          // phi columns (PHI_r 64 | PHI_theta 64)
#define U_   512
#define YST  132          // Y row stride: 128 phi cols + psi_r + psi_theta + 2 pad
#define PI_F 3.14159265358979f

typedef _Float16 v16h __attribute__((ext_vector_type(16)));
typedef _Float16 v8h  __attribute__((ext_vector_type(8)));
typedef float    v8f  __attribute__((ext_vector_type(8)));
typedef float    v4f  __attribute__((ext_vector_type(4)));

// ---------------- kernel W: pack PHI_r|PHI_theta -> f16, [col][K] ----------------
__global__ __launch_bounds__(256) void qs_prep_weights(
    const float* __restrict__ PHI_r, const float* __restrict__ PHI_theta,
    _Float16* __restrict__ PHIt)
{
    int c = blockIdx.x;            // 0..127
    for (int k = threadIdx.x; k < D_; k += 256) {
        float v = (c < NS_) ? PHI_r[k * NS_ + c] : PHI_theta[k * NS_ + (c - NS_)];
        PHIt[c * D_ + k] = (_Float16)v;
    }
}

// ---------------- kernel G: fused WMMA GEMM + psi dot products ----------------
// block = 256 threads (8 waves); each block: 128 rows x 128 cols, K loop of 32.
__global__ __launch_bounds__(256) void qs_gemm(
    const float* __restrict__ x,
    const float* __restrict__ psiRw, const float* __restrict__ psiTw,
    const _Float16* __restrict__ PHIt,
    float* __restrict__ YS)
{
    __shared__ float    sWr[D_];
    __shared__ float    sWt[D_];
    __shared__ _Float16 xt[128][40];   // x tile f16, padded rows (80B)
    __shared__ _Float16 bt[128][40];   // B tile f16, [col][k]
    __shared__ float    psired[512];

    const int t    = threadIdx.x;
    const int b    = blockIdx.x >> 5;
    const int mt   = blockIdx.x & 31;
    const int row0 = mt * 128;

    sWr[t] = psiRw[t];  sWr[t + 256] = psiRw[t + 256];
    sWt[t] = psiTw[t];  sWt[t + 256] = psiTw[t + 256];

    const int wave  = t >> 5;
    const int lane  = t & 31;
    const int laneM = lane & 15;
    const int hiK   = (lane >> 4) * 8;    // A-frag K sub-offset
    const int bK    = (lane >> 4) * 16;   // unused; B frag reads with hiK, same k order as A

    v8f acc[8];
    #pragma unroll
    for (int i = 0; i < 8; ++i)
        #pragma unroll
        for (int j = 0; j < 8; ++j) acc[i][j] = 0.0f;

    float accR = 0.0f, accT = 0.0f;
    const int xr = t >> 1;
    const int xk = (t & 1) * 16;
    const int bc = t >> 1;
    const int bkoff = (t & 1) * 16;

    __syncthreads();

    for (int ks = 0; ks < D_; ks += 32) {
        // --- cooperative x tile load (f32), fused psi accumulation, f16 convert ---
        const float* xrow = x + ((b * L_ + row0 + xr) * D_ + ks + xk);
        float xv[16];
        #pragma unroll
        for (int q = 0; q < 4; ++q) {
            v4f tmp = *(const v4f*)(xrow + 4 * q);
            xv[4*q+0] = tmp[0]; xv[4*q+1] = tmp[1];
            xv[4*q+2] = tmp[2]; xv[4*q+3] = tmp[3];
        }
        #pragma unroll
        for (int j = 0; j < 16; ++j) {
            float v = xv[j];
            accR += v * sWr[ks + xk + j];
            accT += v * sWt[ks + xk + j];
            xt[xr][xk + j] = (_Float16)v;
        }
        // --- cooperative B tile load (already f16, [col][K] contiguous) ---
        const _Float16* bp = PHIt + bc * D_ + ks + bkoff;
        *(v8h*)&bt[bc][bkoff]     = *(const v8h*)(bp);
        *(v8h*)&bt[bc][bkoff + 8] = *(const v8h*)(bp + 8);
        __syncthreads();

        // --- A fragment: 16x32 f16 layout (lanes<16: K {0..7,16..23}; lanes>=16: +8) ---
        const int am = wave * 16 + laneM;
        v8h alo = *(const v8h*)&xt[am][hiK];
        v8h ahi = *(const v8h*)&xt[am][16 + hiK];
        v16h a;
        #pragma unroll
        for (int j = 0; j < 8; ++j) { a[j] = alo[j]; a[8 + j] = ahi[j]; }

        #pragma unroll
        for (int ct = 0; ct < 8; ++ct) {
            const int bn = ct * 16 + laneM;
            v8h blo = *(const v8h*)&bt[bn][hiK];
            v8h bhi = *(const v8h*)&bt[bn][16 + hiK];
            v16h bf;
            #pragma unroll
            for (int j = 0; j < 8; ++j) { bf[j] = blo[j]; bf[8 + j] = bhi[j]; }
            acc[ct] = __builtin_amdgcn_wmma_f32_16x16x32_f16(
                false, a, false, bf, (short)0, acc[ct], false, false);
        }
        __syncthreads();
    }

    // --- psi column writeback (pairwise combine across k-halves) ---
    psired[t] = accR; psired[256 + t] = accT;
    __syncthreads();
    if (t < 128) {
        float r  = psired[2 * t] + psired[2 * t + 1];
        float th = psired[256 + 2 * t] + psired[256 + 2 * t + 1];
        int base = (b * L_ + row0 + t) * YST;
        YS[base + 128] = r;
        YS[base + 129] = th;
    }
    // --- C writeback: VGPR i, lanes 0-15 -> M=i, lanes 16-31 -> M=i+8 ---
    const int rofs = (lane >> 4) * 8;
    #pragma unroll
    for (int ct = 0; ct < 8; ++ct) {
        int col = ct * 16 + laneM;
        #pragma unroll
        for (int i = 0; i < 8; ++i) {
            int row = wave * 16 + i + rofs;
            YS[(b * L_ + row0 + row) * YST + col] = acc[ct][i];
        }
    }
}

// ---------------- psi softmax stats (max + sumexp over L) ----------------
__global__ __launch_bounds__(256) void qs_psi_stats(
    const float* __restrict__ YS, float* __restrict__ psimax, float* __restrict__ psisum)
{
    __shared__ float sm[256], ss[256];
    int b = blockIdx.x, t = threadIdx.x;
    float m = -1e30f;
    for (int l = t; l < L_; l += 256) m = fmaxf(m, YS[(b * L_ + l) * YST + 128]);
    sm[t] = m; __syncthreads();
    for (int st = 128; st > 0; st >>= 1) {
        if (t < st) sm[t] = fmaxf(sm[t], sm[t + st]);
        __syncthreads();
    }
    float M = sm[0];
    float s = 0.0f;
    for (int l = t; l < L_; l += 256) s += __expf(YS[(b * L_ + l) * YST + 128] - M);
    ss[t] = s; __syncthreads();
    for (int st = 128; st > 0; st >>= 1) {
        if (t < st) ss[t] += ss[t + st];
        __syncthreads();
    }
    if (t == 0) { psimax[b] = M; psisum[b] = ss[0]; }
}

// ---------------- phi softmax stats: pass 1 (per 128-row chunk partials) ----------------
__global__ __launch_bounds__(64) void qs_phi_stats1(
    const float* __restrict__ YS, float* __restrict__ Pm, float* __restrict__ Ps)
{
    int b = blockIdx.x >> 5, chunk = blockIdx.x & 31;
    int c = threadIdx.x;                 // 0..63
    int l0 = chunk * 128;
    float m = -1e30f;
    for (int i = 0; i < 128; ++i) m = fmaxf(m, YS[(b * L_ + l0 + i) * YST + c]);
    float s = 0.0f;
    for (int i = 0; i < 128; ++i) s += __expf(YS[(b * L_ + l0 + i) * YST + c] - m);
    int idx = (b * 32 + chunk) * NS_ + c;
    Pm[idx] = m; Ps[idx] = s;
}

// ---------------- phi softmax stats: pass 2 (combine 32 partials) ----------------
__global__ __launch_bounds__(64) void qs_phi_stats2(
    const float* __restrict__ Pm, const float* __restrict__ Ps,
    float* __restrict__ phimax, float* __restrict__ phisum)
{
    int b = blockIdx.x, c = threadIdx.x;
    float m = -1e30f;
    for (int j = 0; j < 32; ++j) m = fmaxf(m, Pm[(b * 32 + j) * NS_ + c]);
    float s = 0.0f;
    for (int j = 0; j < 32; ++j) {
        int idx = (b * 32 + j) * NS_ + c;
        s += Ps[idx] * __expf(Pm[idx] - m);
    }
    phimax[b * NS_ + c] = m;
    phisum[b * NS_ + c] = s;
}

// ---------------- reduce over L: real/imag partial sums ----------------
__global__ __launch_bounds__(256) void qs_reduce_ri(
    const float* __restrict__ YS,
    const float* __restrict__ psimax, const float* __restrict__ phimax,
    const float* __restrict__ psiTbias, const float* __restrict__ phiTbias,
    float* __restrict__ realb, float* __restrict__ imagb)
{
    __shared__ float rre[256], rim[256];
    int b = blockIdx.x >> 3, chunk = blockIdx.x & 7;
    int t = threadIdx.x;
    int n = t & 63, sub = t >> 6;       // 4 L-lanes per state
    float pM  = psimax[b];
    float fM  = phimax[b * NS_ + n];
    float pBias = psiTbias[0];
    float fBias = phiTbias[0];
    float re = 0.0f, im = 0.0f;
    int l0 = chunk * 512 + sub;
    for (int i = 0; i < 128; ++i) {
        int base = (b * L_ + l0 + i * 4) * YST;
        float phiL = YS[base + n];
        float phiT = YS[base + NS_ + n];
        float psiL = YS[base + 128];
        float psiT = YS[base + 129];
        float psiarg = PI_F * (1.0f + tanhf(psiT + pBias));
        float phiarg = -PI_F * (1.0f + PI_F * tanhf(phiT + fBias));
        float e = __expf(0.5f * ((psiL - pM) + (phiL - fM)));
        float sv, cv;
        __sincosf(psiarg - phiarg, &sv, &cv);
        re += e * cv;
        im += e * sv;
    }
    rre[t] = re; rim[t] = im;
    __syncthreads();
    if (t < 64) {
        float R = rre[t] + rre[t + 64] + rre[t + 128] + rre[t + 192];
        float I = rim[t] + rim[t + 64] + rim[t + 128] + rim[t + 192];
        atomicAdd(&realb[b * NS_ + t], R);
        atomicAdd(&imagb[b * NS_ + t], I);
    }
}

// ---------------- final: collapse + [32,64]@[64,512] + bias ----------------
__global__ __launch_bounds__(256) void qs_final(
    const float* __restrict__ realb, const float* __restrict__ imagb,
    const float* __restrict__ psisum, const float* __restrict__ phisum,
    const float* __restrict__ S, const float* __restrict__ Sbias,
    float* __restrict__ out)
{
    __shared__ float col[NS_];
    int b = blockIdx.x, t = threadIdx.x;
    if (t < NS_) {
        float r = realb[b * NS_ + t];
        float i = imagb[b * NS_ + t];
        col[t] = (r * r + i * i) / (psisum[b] * phisum[b * NS_ + t]);
    }
    __syncthreads();
    float bias = Sbias[0];
    for (int u = t; u < U_; u += 256) {
        float accv = bias;
        #pragma unroll 8
        for (int n = 0; n < NS_; ++n) accv += col[n] * S[n * U_ + u];
        out[b * U_ + u] = accv;
    }
}

// ---------------- launcher ----------------
extern "C" void kernel_launch(void* const* d_in, const int* in_sizes, int n_in,
                              void* d_out, int out_size, void* d_ws, size_t ws_size,
                              hipStream_t stream) {
    const float* x        = (const float*)d_in[0];
    const float* PSI_r    = (const float*)d_in[1];
    const float* PSI_t    = (const float*)d_in[2];
    const float* PSI_tb   = (const float*)d_in[3];
    const float* PHI_r    = (const float*)d_in[4];
    const float* PHI_t    = (const float*)d_in[5];
    const float* PHI_tb   = (const float*)d_in[6];
    const float* S        = (const float*)d_in[7];
    const float* S_b      = (const float*)d_in[8];
    float* out = (float*)d_out;

    // workspace carve-up (256B aligned)
    char* p = (char*)d_ws;
    auto take = [&](size_t bytes) -> char* {
        char* r = p;
        p += (bytes + 255) & ~(size_t)255;
        return r;
    };
    float*    YS     = (float*)   take((size_t)B_ * L_ * YST * sizeof(float));  // ~69 MB
    _Float16* PHIt   = (_Float16*)take((size_t)NC_ * D_ * sizeof(_Float16));
    float*    Pm     = (float*)   take((size_t)B_ * 32 * NS_ * sizeof(float));
    float*    Ps     = (float*)   take((size_t)B_ * 32 * NS_ * sizeof(float));
    float*    phimax = (float*)   take((size_t)B_ * NS_ * sizeof(float));
    float*    phisum = (float*)   take((size_t)B_ * NS_ * sizeof(float));
    float*    psimax = (float*)   take((size_t)B_ * sizeof(float));
    float*    psisum = (float*)   take((size_t)B_ * sizeof(float));
    float*    realb  = (float*)   take((size_t)B_ * NS_ * sizeof(float) * 2);   // real|imag contiguous
    float*    imagb  = realb + (size_t)B_ * NS_;

    hipMemsetAsync(realb, 0, (size_t)B_ * NS_ * sizeof(float) * 2, stream);

    qs_prep_weights<<<NC_, 256, 0, stream>>>(PHI_r, PHI_t, PHIt);
    qs_gemm<<<B_ * (L_ / 128), 256, 0, stream>>>(x, PSI_r, PSI_t, PHIt, YS);
    qs_psi_stats<<<B_, 256, 0, stream>>>(YS, psimax, psisum);
    qs_phi_stats1<<<B_ * 32, 64, 0, stream>>>(YS, Pm, Ps);
    qs_phi_stats2<<<B_, 64, 0, stream>>>(Pm, Ps, phimax, phisum);
    qs_reduce_ri<<<B_ * 8, 256, 0, stream>>>(YS, psimax, phimax, PSI_tb, PHI_tb, realb, imagb);
    qs_final<<<B_, 256, 0, stream>>>(realb, imagb, psisum, phisum, S, S_b, out);
}
